// PolicyNetwork_89885075571243
// MI455X (gfx1250) — compile-verified, matrix operand order checked by reading, not run on hardware
//
#include <hip/hip_runtime.h>
#include <hip/hip_bf16.h>
#include <cstdint>
#include <cstddef>

// ---------------------------------------------------------------------------
// Problem constants (match reference)
// ---------------------------------------------------------------------------
#define BB   16
#define NN   1024
#define KK   512
#define DIN  64
#define HH   256
#define NHD  8
#define HD   32
#define LL   2
#define FF   1024
#define MM   (BB * NN)          // 16384 rows through every GEMM

typedef __attribute__((ext_vector_type(16))) _Float16 v16h;
typedef __attribute__((ext_vector_type(8)))  float    v8f;

// ---------------------------------------------------------------------------
// Fragment loader: 16-bit A/B matrix 16x32 (MxK) per CDNA5 ISA 7.12.2.
// Row-major [rows x K]; per-lane row = (lane&15), kbase = (lane<16)?0:8;
// VGPR i in 0..3 -> K = kbase+2i,+1 ; VGPR i in 4..7 -> K = kbase+16+2(i-4),+1.
// Contiguous 16B + 16B at +32B -> compiler emits two global_load_b128.
// ---------------------------------------------------------------------------
__device__ inline v16h load_frag16(const _Float16* __restrict__ p) {
  v16h f;
#pragma unroll
  for (int i = 0; i < 8; ++i) f[i] = p[i];
#pragma unroll
  for (int i = 0; i < 8; ++i) f[8 + i] = p[16 + i];
  return f;
}

#define WMMA8(A0, A1, B0, B1, B2, B3)                                          \
  do {                                                                         \
    acc[0][0] = __builtin_amdgcn_wmma_f32_16x16x32_f16(false, A0, false, B0,   \
                                                       (short)0, acc[0][0], false, false); \
    acc[0][1] = __builtin_amdgcn_wmma_f32_16x16x32_f16(false, A0, false, B1,   \
                                                       (short)0, acc[0][1], false, false); \
    acc[0][2] = __builtin_amdgcn_wmma_f32_16x16x32_f16(false, A0, false, B2,   \
                                                       (short)0, acc[0][2], false, false); \
    acc[0][3] = __builtin_amdgcn_wmma_f32_16x16x32_f16(false, A0, false, B3,   \
                                                       (short)0, acc[0][3], false, false); \
    acc[1][0] = __builtin_amdgcn_wmma_f32_16x16x32_f16(false, A1, false, B0,   \
                                                       (short)0, acc[1][0], false, false); \
    acc[1][1] = __builtin_amdgcn_wmma_f32_16x16x32_f16(false, A1, false, B1,   \
                                                       (short)0, acc[1][1], false, false); \
    acc[1][2] = __builtin_amdgcn_wmma_f32_16x16x32_f16(false, A1, false, B2,   \
                                                       (short)0, acc[1][2], false, false); \
    acc[1][3] = __builtin_amdgcn_wmma_f32_16x16x32_f16(false, A1, false, B3,   \
                                                       (short)0, acc[1][3], false, false); \
  } while (0)

// ---------------------------------------------------------------------------
// Register-blocked, ping-pong double-buffered GEMM.
// C[M,Nn] = A[M,K] (f16) x Bt[Nn,K] (f16 pre-transposed) + bias, opt ReLU.
// One wave computes a 32x64 macro-tile: 2 A-frags x 4 B-frags -> 8 wmmas per
// 32-wide K-step from 6 fragment loads. K is required to be a multiple of 64
// (true for all call sites: 64/256/1024), so the K loop is unrolled 2x with
// two disjoint register sets X/Y -- no rotate-by-copy moves, loads of one set
// overlap the wmma block consuming the other.
// ---------------------------------------------------------------------------
__global__ __launch_bounds__(32) void gemm_f16_wmma(
    const _Float16* __restrict__ A, const _Float16* __restrict__ Bt,
    const float* __restrict__ bias, float* __restrict__ C,
    int M, int Nn, int K, int relu) {
  const int lane = threadIdx.x;
  const int lr = lane & 15;
  const int kb = (lane < 16) ? 0 : 8;
  const int tm = blockIdx.x * 32;
  const int tn = blockIdx.y * 64;

  const _Float16* a0p = A + (size_t)(tm + lr) * K + kb;
  const _Float16* a1p = A + (size_t)(tm + 16 + lr) * K + kb;
  const _Float16* b0p = Bt + (size_t)(tn + 0 * 16 + lr) * K + kb;
  const _Float16* b1p = Bt + (size_t)(tn + 1 * 16 + lr) * K + kb;
  const _Float16* b2p = Bt + (size_t)(tn + 2 * 16 + lr) * K + kb;
  const _Float16* b3p = Bt + (size_t)(tn + 3 * 16 + lr) * K + kb;

  v8f acc[2][4];
#pragma unroll
  for (int i = 0; i < 2; ++i)
#pragma unroll
    for (int j = 0; j < 4; ++j) acc[i][j] = (v8f){};

  // set X: fragments for k0 = 0
  v16h xa0 = load_frag16(a0p), xa1 = load_frag16(a1p);
  v16h xb0 = load_frag16(b0p), xb1 = load_frag16(b1p);
  v16h xb2 = load_frag16(b2p), xb3 = load_frag16(b3p);

  for (int k0 = 0; k0 < K; k0 += 64) {
    // set Y: fragments for k0+32 (always in range since K % 64 == 0)
    const int km = k0 + 32;
    v16h ya0 = load_frag16(a0p + km), ya1 = load_frag16(a1p + km);
    v16h yb0 = load_frag16(b0p + km), yb1 = load_frag16(b1p + km);
    v16h yb2 = load_frag16(b2p + km), yb3 = load_frag16(b3p + km);
    __builtin_prefetch(a0p + km + 32, 0, 1);
    __builtin_prefetch(b0p + km + 32, 0, 1);

    WMMA8(xa0, xa1, xb0, xb1, xb2, xb3);     // consume X while Y in flight

    const int kn = k0 + 64;
    if (kn < K) {                            // refill X for next iteration
      xa0 = load_frag16(a0p + kn);
      xa1 = load_frag16(a1p + kn);
      xb0 = load_frag16(b0p + kn);
      xb1 = load_frag16(b1p + kn);
      xb2 = load_frag16(b2p + kn);
      xb3 = load_frag16(b3p + kn);
      __builtin_prefetch(a0p + kn + 32, 0, 1);
      __builtin_prefetch(b0p + kn + 32, 0, 1);
    }

    WMMA8(ya0, ya1, yb0, yb1, yb2, yb3);     // consume Y while X in flight
  }

  // epilogue: store 2x4 tiles of 16x16
  const int roff = (lane < 16) ? 0 : 8;
#pragma unroll
  for (int ti = 0; ti < 2; ++ti) {
#pragma unroll
    for (int tj = 0; tj < 4; ++tj) {
      const int col = tn + tj * 16 + lr;
      const int rbase = tm + ti * 16 + roff;
      const float bv = bias ? bias[col] : 0.0f;
#pragma unroll
      for (int r = 0; r < 8; ++r) {
        float v = acc[ti][tj][r] + bv;
        if (relu) v = fmaxf(v, 0.0f);
        C[(size_t)(rbase + r) * Nn + col] = v;
      }
    }
  }
}

// ---------------------------------------------------------------------------
// Flash attention: one wave per (16-query tile, head, batch).
// HD == 32 == WMMA K, so QK^T is one wmma per 16-key tile.
// Online softmax with shfl_xor reductions inside 16-lane halves.
// P relayout C-frag -> A-frag through LDS.
// ---------------------------------------------------------------------------
__global__ __launch_bounds__(32) void attn_wmma(
    const float* __restrict__ qkv, float* __restrict__ out) {
  const int lane = threadIdx.x;
  const int qt = blockIdx.x, h = blockIdx.y, b = blockIdx.z;
  const float scale = 0.17677669529663687f;  // 1/sqrt(32)
  const int row3 = 3 * HH;

  __shared__ _Float16 pl[16][36];  // padded 16x32 P tile

  // Q A-fragment (scaled while converting f32->f16)
  const int m = lane & 15;
  const int kb = (lane < 16) ? 0 : 8;
  const float* qrow = qkv + ((size_t)(b * NN + qt * 16 + m)) * row3 + h * HD;
  v16h qa;
#pragma unroll
  for (int i = 0; i < 8; ++i) {
    int hd = kb + ((i < 4) ? 2 * i : 16 + 2 * (i - 4));
    qa[2 * i]     = (_Float16)(qrow[hd] * scale);
    qa[2 * i + 1] = (_Float16)(qrow[hd + 1] * scale);
  }

  v8f acc0 = {}, acc1 = {};
  float mrow[8], lrow[8];
#pragma unroll
  for (int r = 0; r < 8; ++r) { mrow[r] = -3.0e38f; lrow[r] = 0.0f; }

  for (int kblk = 0; kblk < NN; kblk += 32) {
    // --- K^T B-fragments (keys kblk..+15 and kblk+16..+31) ---
    v16h kb0, kb1;
#pragma unroll
    for (int t = 0; t < 2; ++t) {
      const int key = kblk + t * 16 + (lane & 15);
      const float* krow = qkv + ((size_t)(b * NN + key)) * row3 + HH + h * HD;
#pragma unroll
      for (int i = 0; i < 8; ++i) {
        int hd = kb + ((i < 4) ? 2 * i : 16 + 2 * (i - 4));
        _Float16 e0 = (_Float16)krow[hd];
        _Float16 e1 = (_Float16)krow[hd + 1];
        if (t == 0) { kb0[2 * i] = e0; kb0[2 * i + 1] = e1; }
        else        { kb1[2 * i] = e0; kb1[2 * i + 1] = e1; }
      }
    }
    v8f z = {};
    v8f s0 = __builtin_amdgcn_wmma_f32_16x16x32_f16(false, qa, false, kb0,
                                                    (short)0, z, false, false);
    v8f s1 = __builtin_amdgcn_wmma_f32_16x16x32_f16(false, qa, false, kb1,
                                                    (short)0, z, false, false);
    // --- online softmax update (per-row stats live per VGPR slot) ---
    float p0[8], p1[8];
#pragma unroll
    for (int r = 0; r < 8; ++r) {
      float tmax = fmaxf(s0[r], s1[r]);
#pragma unroll
      for (int msk = 1; msk < 16; msk <<= 1)
        tmax = fmaxf(tmax, __shfl_xor(tmax, msk));
      float mnew = fmaxf(mrow[r], tmax);
      float corr = __expf(mrow[r] - mnew);
      p0[r] = __expf(s0[r] - mnew);
      p1[r] = __expf(s1[r] - mnew);
      float rs = p0[r] + p1[r];
#pragma unroll
      for (int msk = 1; msk < 16; msk <<= 1)
        rs += __shfl_xor(rs, msk);
      lrow[r] = lrow[r] * corr + rs;
      mrow[r] = mnew;
      acc0[r] *= corr;
      acc1[r] *= corr;
    }
    // --- P: C-frag layout -> LDS -> A-frag layout ---
    __syncthreads();
    {
      const int roff = (lane < 16) ? 0 : 8;
      const int c = lane & 15;
#pragma unroll
      for (int r = 0; r < 8; ++r) {
        pl[r + roff][c]      = (_Float16)p0[r];
        pl[r + roff][16 + c] = (_Float16)p1[r];
      }
    }
    __syncthreads();
    v16h pa;
#pragma unroll
    for (int i = 0; i < 8; ++i) {
      int k = kb + ((i < 4) ? 2 * i : 16 + 2 * (i - 4));
      pa[2 * i]     = pl[m][k];
      pa[2 * i + 1] = pl[m][k + 1];
    }
    // --- V B-fragments (hd columns 0..15 and 16..31) ---
    v16h vb0, vb1;
#pragma unroll
    for (int t = 0; t < 2; ++t) {
      const int hd = t * 16 + (lane & 15);
#pragma unroll
      for (int i = 0; i < 8; ++i) {
        int k = kb + ((i < 4) ? 2 * i : 16 + 2 * (i - 4));
        const float* v0 = qkv + ((size_t)(b * NN + kblk + k)) * row3 + 2 * HH + h * HD + hd;
        const float* v1 = qkv + ((size_t)(b * NN + kblk + k + 1)) * row3 + 2 * HH + h * HD + hd;
        _Float16 e0 = (_Float16)(*v0);
        _Float16 e1 = (_Float16)(*v1);
        if (t == 0) { vb0[2 * i] = e0; vb0[2 * i + 1] = e1; }
        else        { vb1[2 * i] = e0; vb1[2 * i + 1] = e1; }
      }
    }
    acc0 = __builtin_amdgcn_wmma_f32_16x16x32_f16(false, pa, false, vb0,
                                                  (short)0, acc0, false, false);
    acc1 = __builtin_amdgcn_wmma_f32_16x16x32_f16(false, pa, false, vb1,
                                                  (short)0, acc1, false, false);
  }

  // write O = acc / l   (out layout [B*N, H])
  const int roff = (lane < 16) ? 0 : 8;
  const int c = lane & 15;
#pragma unroll
  for (int r = 0; r < 8; ++r) {
    const size_t row = (size_t)(b * NN + qt * 16 + r + roff);
    float inv = 1.0f / lrow[r];
    out[row * HH + h * HD + c]      = acc0[r] * inv;
    out[row * HH + h * HD + 16 + c] = acc1[r] * inv;
  }
}

// ---------------------------------------------------------------------------
// LayerNorm(residual): out = LN(a + d) * g + b ; one wave per 256-wide row.
// ---------------------------------------------------------------------------
__global__ __launch_bounds__(256) void ln_res(
    const float* __restrict__ a, const float* __restrict__ d,
    const float* __restrict__ g, const float* __restrict__ be,
    float* __restrict__ out, int rows) {
  const int wave = threadIdx.x >> 5;
  const int lane = threadIdx.x & 31;
  const int row = blockIdx.x * 8 + wave;
  if (row >= rows) return;
  const float* pa = a + (size_t)row * HH;
  const float* pd = d + (size_t)row * HH;
  float v[8], s = 0.0f, s2 = 0.0f;
#pragma unroll
  for (int i = 0; i < 8; ++i) {
    int idx = lane + 32 * i;
    float t = pa[idx] + pd[idx];
    v[i] = t; s += t; s2 += t * t;
  }
#pragma unroll
  for (int msk = 1; msk < 32; msk <<= 1) {
    s  += __shfl_xor(s, msk);
    s2 += __shfl_xor(s2, msk);
  }
  const float mean = s * (1.0f / HH);
  const float var  = s2 * (1.0f / HH) - mean * mean;
  const float inv  = rsqrtf(var + 1e-5f);
  float* po = out + (size_t)row * HH;
#pragma unroll
  for (int i = 0; i < 8; ++i) {
    int idx = lane + 32 * i;
    po[idx] = (v[i] - mean) * inv * g[idx] + be[idx];
  }
}

// ---------------------------------------------------------------------------
// Conversion helpers
// ---------------------------------------------------------------------------
__global__ void cvt_f32_f16(const float* __restrict__ src,
                            _Float16* __restrict__ dst, size_t n) {
  for (size_t i = (size_t)blockIdx.x * blockDim.x + threadIdx.x; i < n;
       i += (size_t)gridDim.x * blockDim.x)
    dst[i] = (_Float16)src[i];
}

// W[K][Nn] -> Wt[Nn][K] (f16)
__global__ void cvt_trans_f16(const float* __restrict__ W,
                              _Float16* __restrict__ Wt, int K, int Nn) {
  size_t i = (size_t)blockIdx.x * blockDim.x + threadIdx.x;
  if (i < (size_t)K * Nn) {
    int k = (int)(i / Nn), n = (int)(i % Nn);
    Wt[(size_t)n * K + k] = (_Float16)W[i];
  }
}

__global__ void zero_f32(float* __restrict__ p, size_t n) {
  size_t i = (size_t)blockIdx.x * blockDim.x + threadIdx.x;
  if (i < n) p[i] = 0.0f;
}

__global__ void copy_idx(const int* __restrict__ s, int* __restrict__ d, int n) {
  int i = blockIdx.x * blockDim.x + threadIdx.x;
  if (i < n) d[i] = s[i];
}

// ---------------------------------------------------------------------------
// Allocation head: one block per batch element.
// ---------------------------------------------------------------------------
__global__ __launch_bounds__(256) void alloc_kernel(
    const float* __restrict__ enc, const int* __restrict__ sel,
    const float* __restrict__ Wc, const float* __restrict__ bc,
    float* __restrict__ bw) {
  const int b = blockIdx.x, t = threadIdx.x;
  __shared__ float meanS[HH];
  __shared__ float ctxS[HH];
  __shared__ float scoreS[KK];
  __shared__ float allocS[KK];
  __shared__ float redF[256];
  __shared__ int   redI[256];
  const int* sb = sel + b * KK;

  int c = 0;
  for (int k = t; k < KK; k += 256) c += (sb[k] < NN) ? 1 : 0;
  redI[t] = c; __syncthreads();
  for (int s = 128; s > 0; s >>= 1) { if (t < s) redI[t] += redI[t + s]; __syncthreads(); }
  int cnt = redI[0]; if (cnt < 1) cnt = 1;

  // gather-mean, channel t
  float acc = 0.0f;
  for (int k = 0; k < KK; ++k) {
    int idx = sb[k];
    if (idx < NN) acc += enc[((size_t)b * NN + idx) * HH + t];
  }
  meanS[t] = acc / (float)cnt;
  __syncthreads();

  // ctx = mean @ Wc + bc
  float cv = bc[t];
  for (int hh = 0; hh < HH; ++hh) cv += meanS[hh] * Wc[hh * HH + t];
  ctxS[t] = cv;
  __syncthreads();

  // scores
  for (int k = t; k < KK; k += 256) {
    int idx = sb[k];
    float s = -1e9f;
    if (idx < NN) {
      const float* row = enc + ((size_t)b * NN + idx) * HH;
      s = 0.0f;
      for (int hh = 0; hh < HH; ++hh) s += ctxS[hh] * row[hh];
    }
    scoreS[k] = s;
  }
  __syncthreads();

  // softmax max
  float mx = -1e30f;
  for (int k = t; k < KK; k += 256) mx = fmaxf(mx, scoreS[k]);
  redF[t] = mx; __syncthreads();
  for (int s = 128; s > 0; s >>= 1) { if (t < s) redF[t] = fmaxf(redF[t], redF[t + s]); __syncthreads(); }
  mx = redF[0]; __syncthreads();

  float se = 0.0f;
  for (int k = t; k < KK; k += 256) se += __expf(scoreS[k] - mx);
  redF[t] = se; __syncthreads();
  for (int s = 128; s > 0; s >>= 1) { if (t < s) redF[t] += redF[t + s]; __syncthreads(); }
  const float denom = redF[0]; __syncthreads();

  // rounded allocations
  float asum = 0.0f;
  for (int k = t; k < KK; k += 256) {
    int idx = sb[k];
    float raw = (idx < NN) ? (__expf(scoreS[k] - mx) / denom * 100.0f) : 0.0f;
    float a = rintf(raw);            // round-half-even like jnp.round
    allocS[k] = a; asum += a;
  }
  redF[t] = asum; __syncthreads();
  for (int s = 128; s > 0; s >>= 1) { if (t < s) redF[t] += redF[t + s]; __syncthreads(); }
  if (t == 0) {
    float diff = 100.0f - redF[0];
    if (sb[0] < NN) allocS[0] += diff;
  }
  __syncthreads();

  // scatter
  for (int k = t; k < KK; k += 256) {
    int idx = sb[k];
    if (idx < NN) bw[(size_t)b * NN + idx] = allocS[k];
  }
}

// ---------------------------------------------------------------------------
// Host orchestration
// ---------------------------------------------------------------------------
extern "C" void kernel_launch(void* const* d_in, const int* in_sizes, int n_in,
                              void* d_out, int out_size, void* d_ws, size_t ws_size,
                              hipStream_t stream) {
  (void)in_sizes; (void)n_in; (void)out_size; (void)ws_size;

  const float* x    = (const float*)d_in[0];
  const int*   sel  = (const int*)d_in[1];
  const float* W_in = (const float*)d_in[2];
  const float* b_in = (const float*)d_in[3];
  const float* Wqkv = (const float*)d_in[4];
  const float* bqkv = (const float*)d_in[5];
  const float* Wo   = (const float*)d_in[6];
  const float* bo   = (const float*)d_in[7];
  const float* ln1g = (const float*)d_in[8];
  const float* ln1b = (const float*)d_in[9];
  const float* W1   = (const float*)d_in[10];
  const float* b1   = (const float*)d_in[11];
  const float* W2   = (const float*)d_in[12];
  const float* b2   = (const float*)d_in[13];
  const float* ln2g = (const float*)d_in[14];
  const float* ln2b = (const float*)d_in[15];
  const float* Wc   = (const float*)d_in[16];
  const float* bc   = (const float*)d_in[17];

  uint8_t* ws = (uint8_t*)d_ws;
  const size_t SZ_H   = (size_t)MM * HH * 4;        // 16 MB
  float*    h      = (float*)(ws);
  float*    h2     = (float*)(ws + SZ_H);
  _Float16* act_h  = (_Float16*)(ws + 2 * SZ_H);    // MM*1024 f16 = 32 MB
  uint8_t*  big    = ws + 2 * SZ_H + (size_t)MM * 1024 * 2;
  float*    qkv    = (float*)big;                   // 48 MB
  float*    attn_o = (float*)(big + (size_t)MM * 3 * HH * 4);  // 16 MB
  float*    ff1    = (float*)big;                   // 64 MB (reuses qkv+attn region)
  float*    tmp    = (float*)(big + (size_t)MM * FF * 4);      // 16 MB
  _Float16* wt_h   = (_Float16*)((uint8_t*)tmp + SZ_H);        // <=512 KB

  int*   out_idx = (int*)d_out;
  float* out_bw  = ((float*)d_out) + BB * KK;

  const int CVT = 256;
  auto nblk = [](size_t n, int b) { return (int)((n + b - 1) / b); };

  // outputs: indices passthrough + zeroed bandwidth map
  copy_idx<<<nblk(BB * KK, CVT), CVT, 0, stream>>>(sel, out_idx, BB * KK);
  zero_f32<<<nblk((size_t)BB * NN, CVT), CVT, 0, stream>>>(out_bw, (size_t)BB * NN);

  // input projection: h = x @ W_in + b_in
  cvt_f32_f16<<<4096, CVT, 0, stream>>>(x, act_h, (size_t)MM * DIN);
  cvt_trans_f16<<<nblk((size_t)DIN * HH, CVT), CVT, 0, stream>>>(W_in, wt_h, DIN, HH);
  gemm_f16_wmma<<<dim3(MM / 32, HH / 64), 32, 0, stream>>>(
      act_h, wt_h, b_in, h, MM, HH, DIN, 0);

  for (int l = 0; l < LL; ++l) {
    // qkv = h @ Wqkv + bqkv
    cvt_f32_f16<<<4096, CVT, 0, stream>>>(h, act_h, (size_t)MM * HH);
    cvt_trans_f16<<<nblk((size_t)HH * 3 * HH, CVT), CVT, 0, stream>>>(
        Wqkv + (size_t)l * HH * 3 * HH, wt_h, HH, 3 * HH);
    gemm_f16_wmma<<<dim3(MM / 32, (3 * HH) / 64), 32, 0, stream>>>(
        act_h, wt_h, bqkv + (size_t)l * 3 * HH, qkv, MM, 3 * HH, HH, 0);

    // attention
    attn_wmma<<<dim3(NN / 16, NHD, BB), 32, 0, stream>>>(qkv, attn_o);

    // proj = attn_o @ Wo + bo ; h2 = LN1(h + proj)
    cvt_f32_f16<<<4096, CVT, 0, stream>>>(attn_o, act_h, (size_t)MM * HH);
    cvt_trans_f16<<<nblk((size_t)HH * HH, CVT), CVT, 0, stream>>>(
        Wo + (size_t)l * HH * HH, wt_h, HH, HH);
    gemm_f16_wmma<<<dim3(MM / 32, HH / 64), 32, 0, stream>>>(
        act_h, wt_h, bo + (size_t)l * HH, tmp, MM, HH, HH, 0);
    ln_res<<<MM / 8, 256, 0, stream>>>(h, tmp, ln1g + (size_t)l * HH,
                                       ln1b + (size_t)l * HH, h2, MM);

    // ff1 = relu(h2 @ W1 + b1)
    cvt_f32_f16<<<4096, CVT, 0, stream>>>(h2, act_h, (size_t)MM * HH);
    cvt_trans_f16<<<nblk((size_t)HH * FF, CVT), CVT, 0, stream>>>(
        W1 + (size_t)l * HH * FF, wt_h, HH, FF);
    gemm_f16_wmma<<<dim3(MM / 32, FF / 64), 32, 0, stream>>>(
        act_h, wt_h, b1 + (size_t)l * FF, ff1, MM, FF, HH, 1);

    // ff2 = ff1 @ W2 + b2 ; h = LN2(h2 + ff2)
    cvt_f32_f16<<<8192, CVT, 0, stream>>>(ff1, act_h, (size_t)MM * FF);
    cvt_trans_f16<<<nblk((size_t)FF * HH, CVT), CVT, 0, stream>>>(
        W2 + (size_t)l * FF * HH, wt_h, FF, HH);
    gemm_f16_wmma<<<dim3(MM / 32, HH / 64), 32, 0, stream>>>(
        act_h, wt_h, b2 + (size_t)l * HH, tmp, MM, HH, FF, 0);
    ln_res<<<MM / 8, 256, 0, stream>>>(h2, tmp, ln2g + (size_t)l * HH,
                                       ln2b + (size_t)l * HH, h, MM);
  }

  // allocation head
  alloc_kernel<<<BB, 256, 0, stream>>>(h, sel, Wc, bc, out_bw);
}
